// LSTM_73882027425815
// MI455X (gfx1250) — compile-verified, matrix operand order checked by reading, not run on hardware
//
#include <hip/hip_runtime.h>

// ---------------------------------------------------------------------------
// LSTM for MI455X (gfx1250, wave32, WMMA).
//   B=64, T=2048, I=H=512.  out = (h_T, c_T) fp32, 2*64*512 elements.
//
// Phase 0 (prep):   transpose+convert weights to bf16 [N,K] layout, bias sums.
// Phase 1 (xproj):  xs = x @ [Wi|Wf|Wg|Wo] + (bi+bh), computed on tiles of
//                   (16 batches x fixed t) and stored in WMMA *fragment order*
//                   so the scan can read each 16x16 C-tile as one b128/lane.
//                   B-fragments batch-loaded from LDS; A loads software-
//                   pipelined in fp32 registers (conversion co-executes with
//                   the XDL WMMAs).  One 16x64 strip per wave -> no spills.
// Phase 2 (scan):   16 persistent WGs, Wh gate slice (128KB) cached in LDS
//                   (CDNA5 WGP has 320KB), c in registers across all 2048
//                   steps, h double-buffered bf16 in global, atomic-counter
//                   grid sync per step, A-fragment prefetch + batched B
//                   fragments in the K-loop.
// ---------------------------------------------------------------------------

#define BB   64
#define TT   2048
#define II   512
#define HH   512
#define G4   2048          // 4*H
#define NWG  16            // persistent workgroups in scan
#define HCOL 32            // h-columns owned per scan workgroup

typedef __bf16          v16bf  __attribute__((ext_vector_type(16)));
typedef float           v8f    __attribute__((ext_vector_type(8)));
typedef float           f32x4  __attribute__((ext_vector_type(4)));
typedef unsigned short  u16x8  __attribute__((ext_vector_type(8)));
typedef unsigned short  u16x16 __attribute__((ext_vector_type(16)));

__device__ __forceinline__ unsigned short f2bf(float f) {
    unsigned u = __builtin_bit_cast(unsigned, f);
    u += 0x7fffu + ((u >> 16) & 1u);          // round-to-nearest-even
    return (unsigned short)(u >> 16);
}
__device__ __forceinline__ float bf2f(unsigned short s) {
    unsigned u = ((unsigned)s) << 16;
    return __builtin_bit_cast(float, u);
}

// One 16-bit WMMA fragment for this lane: per ISA 7.12.2 the lane's 16 halves
// are two contiguous 16-byte chunks at +0 and +32 bytes (K.. and K+16..).
__device__ __forceinline__ v16bf load_frag(const unsigned short* p) {
    u16x8 lo = *(const u16x8*)(p);
    u16x8 hi = *(const u16x8*)(p + 16);
    u16x16 u = __builtin_shufflevector(lo, hi, 0,1,2,3,4,5,6,7,8,9,10,11,12,13,14,15);
    return __builtin_bit_cast(v16bf, u);
}

__device__ __forceinline__ v8f wmma_bf16(v16bf a, v16bf b, v8f c) {
    // (neg_a, A, neg_b, B, c_mod, C, reuse_a, reuse_b)
    return __builtin_amdgcn_wmma_f32_16x16x32_bf16(false, a, false, b, (short)0, c, false, false);
}

__device__ __forceinline__ float fsig(float x)  { return 1.0f / (1.0f + __expf(-x)); }
__device__ __forceinline__ float ftanh(float x) { return 1.0f - 2.0f / (__expf(2.0f * x) + 1.0f); }

// Fragment-ordered xs:  [t][strip(4)][colblock(128)][lane(32)][r(8)] bf16.
// strip = batch/16, colblock = (4H column)/16.  Element (lane, r) of a tile is
// D-layout: row = 16*strip + 8*(lane>>4) + r, col = 16*colblock + (lane&15).
__device__ __forceinline__ size_t xs_idx(int t, int s, int cb, int lane) {
    return ((((size_t)t * 4 + s) * 128 + cb) * 32 + lane) * 8;
}

// ---------------------------------------------------------------------------
// Phase 0: weights -> bf16, transposed to [N=4H, K] so B-fragments are two
// contiguous b128 loads per lane.  bsum = b_i* + b_h*.
// ---------------------------------------------------------------------------
__global__ void k_prep(const float* __restrict__ W_ii, const float* __restrict__ W_hi,
                       const float* __restrict__ W_if, const float* __restrict__ W_hf,
                       const float* __restrict__ W_ig, const float* __restrict__ W_hg,
                       const float* __restrict__ W_io, const float* __restrict__ W_ho,
                       const float* __restrict__ b_ii, const float* __restrict__ b_hi,
                       const float* __restrict__ b_if, const float* __restrict__ b_hf,
                       const float* __restrict__ b_ig, const float* __restrict__ b_hg,
                       const float* __restrict__ b_io, const float* __restrict__ b_ho,
                       unsigned short* __restrict__ WiT, unsigned short* __restrict__ WhT,
                       float* __restrict__ bsum)
{
    const int n  = blockIdx.x;          // 0..2047 = gate*512 + n_local
    const int g  = n >> 9;
    const int nl = n & 511;
    const float* wi = (g == 0) ? W_ii : (g == 1) ? W_if : (g == 2) ? W_ig : W_io;
    const float* wh = (g == 0) ? W_hi : (g == 1) ? W_hf : (g == 2) ? W_hg : W_ho;
    const float* bi = (g == 0) ? b_ii : (g == 1) ? b_if : (g == 2) ? b_ig : b_io;
    const float* bh = (g == 0) ? b_hi : (g == 1) ? b_hf : (g == 2) ? b_hg : b_ho;
    for (int k = threadIdx.x; k < II; k += blockDim.x) {
        WiT[(size_t)n * II + k] = f2bf(wi[(size_t)k * HH + nl]);   // W is [K, H] row-major
        WhT[(size_t)n * HH + k] = f2bf(wh[(size_t)k * HH + nl]);
    }
    if (threadIdx.x == 0) bsum[n] = bi[nl] + bh[nl];
}

// ---------------------------------------------------------------------------
// Phase 1: xs = x @ Wi + bsum, fragment-ordered output.
// Tile M-dim = 16 batches at fixed t (A row b of tile (t,strip) is
// x[(16*strip+b)*T + t, :]).  Each wave: one (t, strip), 4 N-tiles (16x64),
// K=512.  WiT chunk (64 cols x 512 K bf16 = 64KB) staged in LDS per block.
// Inner loop: batch all 8 ds_load_b128 (4 B frags) before the 4 WMMAs so
// the waits drain progressively instead of load->wait0->wmma serialization.
// VGPR budget: acc(32) + bF(32) + cur/nxt(32) + aF(8) ~ 120 -> no spills.
// grid: 1024 tgroups x 32 nchunks (nchunk fastest for L2 reuse of x rows),
// blockDim 256 (8 waves = 2 t x 4 strips).
// ---------------------------------------------------------------------------
__global__ void k_xproj(const float* __restrict__ x,
                        const unsigned short* __restrict__ WiT,
                        const float* __restrict__ bsum,
                        unsigned short* __restrict__ xs)
{
    extern __shared__ unsigned short smem[];          // 64*512 bf16 = 64KB

    const int tid  = threadIdx.x;
    const int lane = tid & 31;
    const int w    = tid >> 5;
    const int lm   = lane & 15;
    const int half = lane >> 4;

    const int nchunk = blockIdx.x & 31;
    const int tgroup = blockIdx.x >> 5;               // 0..1023
    const int c0     = nchunk * 64;

    // Stage WiT rows [c0, c0+64) into LDS (contiguous 64KB).
    {
        const unsigned short* src = WiT + (size_t)c0 * II;
        for (int i = tid; i < 64 * II / 8; i += 256)
            *(u16x8*)(smem + i * 8) = *(const u16x8*)(src + i * 8);
    }
    __syncthreads();

    const int s = w & 3;                              // batch strip 0..3
    const int t = tgroup * 2 + (w >> 2);              // 0..2047
    const float* arow = x + ((size_t)(s * 16 + lm) * TT + t) * II;

    v8f acc[4] = {};
    f32x4 cur[4], nxt[4];
    {                                                 // prime the pipeline (ks=0)
        const int ko = half * 8;
        cur[0] = *(const f32x4*)(arow + ko);
        cur[1] = *(const f32x4*)(arow + ko + 4);
        cur[2] = *(const f32x4*)(arow + ko + 16);
        cur[3] = *(const f32x4*)(arow + ko + 20);
    }

#pragma unroll
    for (int ks = 0; ks < 16; ++ks) {
        if (ks < 15) {                                // A loads for ks+1 in flight
            const int ko = (ks + 1) * 32 + half * 8;
            nxt[0] = *(const f32x4*)(arow + ko);
            nxt[1] = *(const f32x4*)(arow + ko + 4);
            nxt[2] = *(const f32x4*)(arow + ko + 16);
            nxt[3] = *(const f32x4*)(arow + ko + 20);
        }
        const int ko = ks * 32 + half * 8;
        v16bf bF[4];                                  // batch all B ds loads first
#pragma unroll
        for (int jt = 0; jt < 4; ++jt)
            bF[jt] = load_frag(smem + (size_t)(jt * 16 + lm) * II + ko);
        u16x16 ua;                                    // fp32 -> bf16 (RNE)
#pragma unroll
        for (int i = 0; i < 4; ++i) {
            ua[i]      = f2bf(cur[0][i]);
            ua[4 + i]  = f2bf(cur[1][i]);
            ua[8 + i]  = f2bf(cur[2][i]);
            ua[12 + i] = f2bf(cur[3][i]);
        }
        const v16bf aF = __builtin_bit_cast(v16bf, ua);
#pragma unroll
        for (int jt = 0; jt < 4; ++jt)
            acc[jt] = wmma_bf16(aF, bF[jt], acc[jt]);
#pragma unroll
        for (int i = 0; i < 4; ++i)
            cur[i] = nxt[i];
    }

    // Epilogue: + bias, pack 8 rows per lane, one b128 store per tile
    // (lanes are consecutive 16B -> 512B fully-coalesced per tile).
#pragma unroll
    for (int jt = 0; jt < 4; ++jt) {
        const int cb  = nchunk * 4 + jt;
        const float bv = bsum[c0 + jt * 16 + lm];
        u16x8 o;
#pragma unroll
        for (int r = 0; r < 8; ++r)
            o[r] = f2bf(acc[jt][r] + bv);
        *(u16x8*)(xs + xs_idx(t, s, cb, lane)) = o;
    }
}

// ---------------------------------------------------------------------------
// Phase 2: persistent scan.  16 WGs x 256 threads.  WG wg owns h columns
// [wg*32, wg*32+32); its WhT slice (4 gates x 32 cols x 512 K bf16 = 128KB)
// lives in LDS.  Wave (s, ch) computes the 16x16 tile at batch strip s,
// col half ch for all 4 gates.  xs tiles arrive as 4 b128 loads issued at
// step top and consumed only in the epilogue (acc starts at 0), fully
// overlapping the K-loop.  A (h) fragment prefetched one k-step ahead; the
// 4 gate B fragments are batch-loaded from LDS before the 4 WMMAs.
// ---------------------------------------------------------------------------
__global__ void k_scan(const unsigned short* __restrict__ xs,
                       const unsigned short* __restrict__ WhT,
                       unsigned short* __restrict__ h0buf,
                       unsigned short* __restrict__ h1buf,
                       float* __restrict__ out,
                       int* __restrict__ syncc)
{
    extern __shared__ unsigned short smem[];          // [128][512] bf16 = 128KB

    const int tid  = threadIdx.x;
    const int lane = tid & 31;
    const int w    = tid >> 5;
    const int lm   = lane & 15;
    const int half = lane >> 4;
    const int wg   = blockIdx.x;                      // 0..15

    // Stage WhT slice: 4 contiguous 32KB chunks (rows g*512 + wg*32 .. +32).
    for (int g = 0; g < 4; ++g) {
        const unsigned short* src = WhT + (size_t)(g * HH + wg * HCOL) * HH;
        unsigned short* dst = smem + g * HCOL * HH;
        for (int i = tid; i < HCOL * HH / 8; i += 256)
            *(u16x8*)(dst + i * 8) = *(const u16x8*)(src + i * 8);
    }
    __syncthreads();

    const int s  = w & 3;                 // batch strip: rows m0..m0+15
    const int ch = w >> 2;                // column half within the WG slice
    const int m0 = s * 16;
    const int j0 = wg * HCOL + ch * 16;   // global h column base of this tile

    v8f c = {};                           // c tile, in VGPRs for all 2048 steps
    v8f hlast = {};
    unsigned short* hb[2] = { h0buf, h1buf };

    for (int t = 0; t < TT; ++t) {
        const unsigned short* hread  = hb[t & 1];
        unsigned short*       hwrite = hb[(t + 1) & 1];

        // xs tiles for this step: one b128 per gate, used only in epilogue.
        u16x8 xsu[4];
#pragma unroll
        for (int g = 0; g < 4; ++g)
            xsu[g] = *(const u16x8*)(xs + xs_idx(t, s, g * 32 + wg * 2 + ch, lane));

        // gates = h @ Wh   (K = 512, 16 chained WMMAs per gate tile).
        v8f acc[4] = {};
        const unsigned short* hrow = hread + (size_t)(m0 + lm) * HH;
        v16bf aC = load_frag(hrow + half * 8);
#pragma unroll
        for (int ks = 0; ks < 16; ++ks) {
            v16bf aN;
            if (ks < 15) aN = load_frag(hrow + (ks + 1) * 32 + half * 8);
            v16bf bF[4];                  // batch the 4 gate B fragments
#pragma unroll
            for (int g = 0; g < 4; ++g)
                bF[g] = load_frag(smem + (size_t)(g * HCOL + ch * 16 + lm) * HH
                                       + ks * 32 + half * 8);
#pragma unroll
            for (int g = 0; g < 4; ++g)
                acc[g] = wmma_bf16(aC, bF[g], acc[g]);
            aC = aN;
        }

        // Elementwise LSTM cell update (xs + bias folded in here).
#pragma unroll
        for (int r = 0; r < 8; ++r) {
            const float iv = fsig (acc[0][r] + bf2f(xsu[0][r]));
            const float fv = fsig (acc[1][r] + bf2f(xsu[1][r]));
            const float gv = ftanh(acc[2][r] + bf2f(xsu[2][r]));
            const float ov = fsig (acc[3][r] + bf2f(xsu[3][r]));
            const float cn = fv * c[r] + iv * gv;
            c[r] = cn;
            const float hv = ov * ftanh(cn);
            hlast[r] = hv;
            hwrite[(size_t)(m0 + half * 8 + r) * HH + j0 + lm] = f2bf(hv);
        }

        // Grid-wide step barrier: monotonically increasing counter.
        __threadfence();
        __syncthreads();
        if (tid == 0) {
            __hip_atomic_fetch_add(syncc, 1, __ATOMIC_ACQ_REL, __HIP_MEMORY_SCOPE_AGENT);
            while (__hip_atomic_load(syncc, __ATOMIC_ACQUIRE, __HIP_MEMORY_SCOPE_AGENT)
                   < NWG * (t + 1))
                __builtin_amdgcn_s_sleep(1);
        }
        __syncthreads();
        __threadfence();
    }

    // out = (h_T, c_T) fp32, each [64, 512]
#pragma unroll
    for (int r = 0; r < 8; ++r) {
        const int m = m0 + half * 8 + r;
        const int j = j0 + lm;
        out[(size_t)m * HH + j]           = hlast[r];
        out[BB * HH + (size_t)m * HH + j] = c[r];
    }
}

// ---------------------------------------------------------------------------
extern "C" void kernel_launch(void* const* d_in, const int* in_sizes, int n_in,
                              void* d_out, int out_size, void* d_ws, size_t ws_size,
                              hipStream_t stream)
{
    (void)in_sizes; (void)n_in; (void)out_size; (void)ws_size;

    const float* x    = (const float*)d_in[0];
    const float* W_ii = (const float*)d_in[1];
    const float* W_hi = (const float*)d_in[2];
    const float* W_if = (const float*)d_in[3];
    const float* W_hf = (const float*)d_in[4];
    const float* W_ig = (const float*)d_in[5];
    const float* W_hg = (const float*)d_in[6];
    const float* W_io = (const float*)d_in[7];
    const float* W_ho = (const float*)d_in[8];
    const float* b_ii = (const float*)d_in[9];
    const float* b_hi = (const float*)d_in[10];
    const float* b_if = (const float*)d_in[11];
    const float* b_hf = (const float*)d_in[12];
    const float* b_ig = (const float*)d_in[13];
    const float* b_hg = (const float*)d_in[14];
    const float* b_io = (const float*)d_in[15];
    const float* b_ho = (const float*)d_in[16];

    // Workspace layout
    char* ws = (char*)d_ws;
    size_t o = 0;
    unsigned short* xs  = (unsigned short*)(ws + o); o += (size_t)BB * TT * G4 * 2; // 512 MB
    unsigned short* WiT = (unsigned short*)(ws + o); o += (size_t)G4 * II * 2;      // 2 MB
    unsigned short* WhT = (unsigned short*)(ws + o); o += (size_t)G4 * HH * 2;      // 2 MB
    float*          bsm = (float*)(ws + o);          o += (size_t)G4 * 4;
    unsigned short* h0  = (unsigned short*)(ws + o); o += (size_t)BB * HH * 2;
    unsigned short* h1  = (unsigned short*)(ws + o); o += (size_t)BB * HH * 2;
    int*            syc = (int*)(ws + o);            o += 256;

    hipMemsetAsync(h0, 0, (size_t)BB * HH * 2, stream);   // h(-1) = 0
    hipMemsetAsync(syc, 0, sizeof(int), stream);

    k_prep<<<G4, 256, 0, stream>>>(W_ii, W_hi, W_if, W_hf, W_ig, W_hg, W_io, W_ho,
                                   b_ii, b_hi, b_if, b_hf, b_ig, b_hg, b_io, b_ho,
                                   WiT, WhT, bsm);

    // 1024 tgroups x 32 nchunks; 8 waves/block = 2 t x 4 strips; 64KB LDS.
    k_xproj<<<1024 * 32, 256, 64 * 1024, stream>>>(x, WiT, bsm, xs);

    // Persistent scan: 16 WGs, 128KB dynamic LDS each (CDNA5: 320KB/WGP).
    k_scan<<<NWG, 256, 128 * 1024, stream>>>(xs, WhT, h0, h1, (float*)d_out, syc);
}
